// LlamaAttention_6330781794810
// MI455X (gfx1250) — matrix operand order repaired and checked by reading, not compiled, not run on hardware
//
#include <hip/hip_runtime.h>
#include <hip/hip_bf16.h>

#define S_LEN 2048
#define HID_DIM 4096
#define NH 32
#define NKV 8
#define HD 128

typedef __bf16 bf16_t;
typedef __bf16 v16bf __attribute__((ext_vector_type(16)));
typedef float  v8f   __attribute__((ext_vector_type(8)));

// ---- WMMA fragment loaders (layouts per CDNA5 ISA 7.12.2) ----
// A (16x32 bf16): lane m=lane&15 is row; lanes 0-15 hold K 0-7 & 16-23,
// lanes 16-31 hold K 8-15 & 24-31 (kh = lane>>4 selects half).
__device__ __forceinline__ v16bf ld_fragA(const bf16_t* p, int kh) {
  union { v16bf v; uint4 q[2]; } u;
  u.q[0] = *reinterpret_cast<const uint4*>(p + kh * 8);
  u.q[1] = *reinterpret_cast<const uint4*>(p + 16 + kh * 8);
  return u.v;
}
// B (32x16 bf16), p points at row n of B^T (contiguous K):
// lanes 0-15 hold K 0-7 & 16-23, lanes 16-31 hold K 8-15 & 24-31.
__device__ __forceinline__ v16bf ld_fragB(const bf16_t* p, int kh) {
  union { v16bf v; uint4 q[2]; } u;
  u.q[0] = *reinterpret_cast<const uint4*>(p + kh * 8);
  u.q[1] = *reinterpret_cast<const uint4*>(p + 16 + kh * 8);
  return u.v;
}
#define WMMA_BF16(a, b, c) \
  __builtin_amdgcn_wmma_f32_16x16x32_bf16(false, (a), false, (b), (short)0, (c), false, false)

// ---- fp32 -> bf16 bulk convert (float4 in, 4x bf16 out) ----
__global__ void cvt_f32_bf16(const float* __restrict__ x, bf16_t* __restrict__ y, int n4) {
  int i = blockIdx.x * blockDim.x + threadIdx.x;
  if (i >= n4) return;
  float4 v = reinterpret_cast<const float4*>(x)[i];
  union { bf16_t b[4]; uint2 u; } o;
  o.b[0] = (bf16_t)v.x; o.b[1] = (bf16_t)v.y; o.b[2] = (bf16_t)v.z; o.b[3] = (bf16_t)v.w;
  reinterpret_cast<uint2*>(y)[i] = o.u;
}

// ---- fp32 -> bf16 transpose-convert: y[c][r] = (bf16)x[r][c], 32x32 LDS tile ----
// grid = (C/32, R/32), 256 threads.
__global__ __launch_bounds__(256) void cvt_t_f32_bf16(const float* __restrict__ x,
                                                      bf16_t* __restrict__ y, int R, int C) {
  __shared__ float tile[32][33];
  const int bc = blockIdx.x * 32, br = blockIdx.y * 32;
  const int tx = threadIdx.x & 31, ty = threadIdx.x >> 5;   // 32 x 8
  #pragma unroll
  for (int i = 0; i < 4; ++i)
    tile[ty + i * 8][tx] = x[(size_t)(br + ty + i * 8) * C + bc + tx];
  __syncthreads();
  #pragma unroll
  for (int i = 0; i < 4; ++i)
    y[(size_t)(bc + ty + i * 8) * R + br + tx] = (bf16_t)tile[tx][ty + i * 8];
}

// ---- RoPE fused with bf16 convert. X,Y: [S, nheads*HD]; one thread per (s,h,d<64) pair
__global__ void rope_to_bf16(const float* __restrict__ X, const float* __restrict__ cosT,
                             const float* __restrict__ sinT, bf16_t* __restrict__ Y,
                             int nheads) {
  int idx = blockIdx.x * blockDim.x + threadIdx.x;
  int d = idx & 63;
  int h = (idx >> 6) & (nheads - 1);   // nheads is a power of two (32 or 8)
  int s = idx / (nheads * 64);
  const float* x = X + ((size_t)s * nheads + h) * HD;
  bf16_t*      y = Y + ((size_t)s * nheads + h) * HD;
  float x1 = x[d], x2 = x[d + 64];
  float c1 = cosT[s * HD + d],      s1 = sinT[s * HD + d];
  float c2 = cosT[s * HD + d + 64], s2 = sinT[s * HD + d + 64];
  y[d]      = (bf16_t)(x1 * c1 - x2 * s1);
  y[d + 64] = (bf16_t)(x2 * c2 + x1 * s2);
}

// ---- bf16 GEMM: C[M,N] f32 = A[M,K] * B[K,N], with B supplied pre-transposed BT[N][K] ----
// 128x256 block tile, BK=32, 8 waves each owning a 64x64 sub-tile:
// per K-step per wave: 4 B-frags live + 4 streamed A-frags -> 16 WMMAs / 16 ds_load_b128.
// No register-carried prefetch (it spilled at the ~192-VGPR cap): fetch+stage is fused
// with short-lived temps, and the NEXT tile is warmed with global_prefetch_b8 (address
// regs only, zero data VGPRs), so the next iteration's loads are cache hits.
__global__ __launch_bounds__(256) void gemm_bf16_f32(
    const bf16_t* __restrict__ A, const bf16_t* __restrict__ BT,
    float* __restrict__ C, int M, int N, int K) {
  constexpr int LDA = 40, LDB = 40;       // +8 bf16 pad: keeps 16B align, shifts banks
  __shared__ bf16_t sA[128 * LDA];        // [m][k]
  __shared__ bf16_t sB[256 * LDB];        // [n][k]
  const int tid = threadIdx.x;
  const int lane = tid & 31, wave = tid >> 5;
  const int m16 = lane & 15, kh = lane >> 4;
  const int wm = (wave >> 2) * 64, wn = (wave & 3) * 64;
  const int bm = blockIdx.y * 128, bn = blockIdx.x * 256;

  auto load_tile = [&](int k0) {          // global -> LDS, temps die immediately
    #pragma unroll
    for (int t = 0; t < 2; ++t) {         // A tile: 128 rows x 4 uint4
      int i = tid * 2 + t;
      int r = i >> 2, c4 = i & 3;
      uint4 v = *(reinterpret_cast<const uint4*>(A + (size_t)(bm + r) * K + k0) + c4);
      *reinterpret_cast<uint4*>(&sA[r * LDA + c4 * 8]) = v;
    }
    #pragma unroll
    for (int t = 0; t < 4; ++t) {         // BT tile: 256 rows x 4 uint4
      int i = tid + t * 256;
      int n = i >> 2, c4 = i & 3;
      uint4 v = *(reinterpret_cast<const uint4*>(BT + (size_t)(bn + n) * K + k0) + c4);
      *reinterpret_cast<uint4*>(&sB[n * LDB + c4 * 8]) = v;
    }
  };
  auto prefetch_tile = [&](int k0) {      // warm next tile: no data VGPRs consumed
    #pragma unroll
    for (int t = 0; t < 2; ++t) {
      int i = tid * 2 + t;
      int r = i >> 2, c4 = i & 3;
      __builtin_prefetch(A + (size_t)(bm + r) * K + k0 + c4 * 8, 0, 1);
    }
    #pragma unroll
    for (int t = 0; t < 4; ++t) {
      int i = tid + t * 256;
      int n = i >> 2, c4 = i & 3;
      __builtin_prefetch(BT + (size_t)(bn + n) * K + k0 + c4 * 8, 0, 1);
    }
  };

  v8f acc[4][4];
  #pragma unroll
  for (int i = 0; i < 4; ++i)
    #pragma unroll
    for (int j = 0; j < 4; ++j) acc[i][j] = {};

  for (int k0 = 0; k0 < K; k0 += 32) {
    load_tile(k0);
    if (k0 + 32 < K) prefetch_tile(k0 + 32);
    __syncthreads();
    v16bf bfr[4];
    #pragma unroll
    for (int j = 0; j < 4; ++j)
      bfr[j] = ld_fragB(&sB[(wn + j * 16 + m16) * LDB], kh);
    #pragma unroll
    for (int i = 0; i < 4; ++i) {
      v16bf afr = ld_fragA(&sA[(wm + i * 16 + m16) * LDA], kh);
      #pragma unroll
      for (int j = 0; j < 4; ++j)
        acc[i][j] = WMMA_BF16(afr, bfr[j], acc[i][j]);
    }
    __syncthreads();
  }
  // C/D layout: vgpr r -> row r + 8*kh, col = lane&15
  #pragma unroll
  for (int i = 0; i < 4; ++i)
    #pragma unroll
    for (int j = 0; j < 4; ++j) {
      int row = bm + wm + i * 16 + 8 * kh;
      int col = bn + wn + j * 16 + m16;
      #pragma unroll
      for (int r = 0; r < 8; ++r)
        C[(size_t)(row + r) * N + col] = acc[i][j][r];
    }
}

// ---- causal flash attention, one block = (head, 128 queries), 8 waves x 16 q ----
// V supplied pre-transposed Vt[NKV*HD][S] so LDS staging is contiguous b128.
__global__ __launch_bounds__(256) void attn_fwd(
    const bf16_t* __restrict__ Qb,   // [S, NH*HD] (RoPE'd)
    const bf16_t* __restrict__ Kb,   // [S, NKV*HD] (RoPE'd)
    const bf16_t* __restrict__ Vt,   // [NKV*HD, S]
    bf16_t* __restrict__ Ob) {       // [S, NH*HD]
  constexpr int LDK = 136, LDV = 72, LDP = 72;
  __shared__ bf16_t sK[64 * LDK];        // [key][d]   (row-major == B^T for QK^T)
  __shared__ bf16_t sV[128 * LDV];       // [d][key]   (== B^T for PV)
  __shared__ bf16_t sP[8 * 16 * LDP];    // per-wave P tile [q][key]
  const int h = blockIdx.y, qb = blockIdx.x;
  const int tid = threadIdx.x, lane = tid & 31, wave = tid >> 5;
  const int m16 = lane & 15, kh = lane >> 4;
  const int qbase = qb * 128 + wave * 16;
  const int qbase_s = __builtin_amdgcn_readfirstlane(qbase);  // scalar branch -> EXEC all-1 for WMMA
  const int kvh = h >> 2;                // GQA: 4 Q heads share one KV head
  bf16_t* sPw = sP + wave * 16 * LDP;
  const float scale = 0.08838834764831845f;  // 1/sqrt(128)

  // Q fragments (held in VGPRs for the whole kernel): 4 chunks of d
  v16bf qf[4];
  const bf16_t* qrow = Qb + (size_t)(qbase + m16) * (NH * HD) + h * HD;
  #pragma unroll
  for (int c = 0; c < 4; ++c) qf[c] = ld_fragA(qrow + c * 32, kh);

  v8f o[8];
  #pragma unroll
  for (int t = 0; t < 8; ++t) o[t] = {};
  float rmax[8], rsum[8];
  #pragma unroll
  for (int r = 0; r < 8; ++r) { rmax[r] = -3.0e38f; rsum[r] = 0.f; }

  const int nkb = 2 * (qb + 1);          // key blocks of 64 covering keys <= qb*128+127
  for (int kb = 0; kb < nkb; ++kb) {
    const int k0 = kb * 64;
    // stage K: 64 keys x 128 d = 1024 uint4 (row-major), 4 per thread
    #pragma unroll
    for (int t = 0; t < 4; ++t) {
      int i = tid + t * 256;
      int key = i >> 4, c4 = i & 15;
      *reinterpret_cast<uint4*>(&sK[key * LDK + c4 * 8]) =
          *(reinterpret_cast<const uint4*>(
              Kb + (size_t)(k0 + key) * (NKV * HD) + kvh * HD) + c4);
    }
    // stage V^T: 128 d-rows x 64 keys = 1024 uint4 (contiguous keys), 4 per thread
    #pragma unroll
    for (int t = 0; t < 4; ++t) {
      int i = tid + t * 256;
      int d = i >> 3, c4 = i & 7;
      *reinterpret_cast<uint4*>(&sV[d * LDV + c4 * 8]) =
          *(reinterpret_cast<const uint4*>(
              Vt + (size_t)(kvh * HD + d) * S_LEN + k0) + c4);
    }
    // prefetch next key block into cache (global_prefetch_b8)
    if (kb + 1 < nkb) {
      __builtin_prefetch(Kb + (size_t)(k0 + 64 + (tid >> 2)) * (NKV * HD) + kvh * HD +
                             (tid & 3) * 32, 0, 1);
      __builtin_prefetch(Vt + (size_t)(kvh * HD + (tid >> 1)) * S_LEN + k0 + 64 +
                             (tid & 1) * 32, 0, 1);
    }
    __syncthreads();
    if (k0 <= qbase_s + 15) {            // wave-uniform causal block skip
      // S = Q K^T  (4 key sub-tiles x 4 d-chunks)
      v8f s4[4];
      #pragma unroll
      for (int nt = 0; nt < 4; ++nt) {
        v8f a = {};
        #pragma unroll
        for (int c = 0; c < 4; ++c) {
          v16bf bk = ld_fragB(&sK[(nt * 16 + m16) * LDK + c * 32], kh);
          a = WMMA_BF16(qf[c], bk, a);
        }
        s4[nt] = a;
      }
      // scale + causal mask (mask value matches reference -1e9)
      #pragma unroll
      for (int nt = 0; nt < 4; ++nt) {
        int keyg = k0 + nt * 16 + m16;
        #pragma unroll
        for (int r = 0; r < 8; ++r) {
          int qg = qbase + r + 8 * kh;
          s4[nt][r] = s4[nt][r] * scale + (keyg <= qg ? 0.f : -1.0e9f);
        }
      }
      // row max (reduce across the 16 lanes of each half-wave group)
      float mnew[8];
      #pragma unroll
      for (int r = 0; r < 8; ++r) {
        float m = fmaxf(fmaxf(s4[0][r], s4[1][r]), fmaxf(s4[2][r], s4[3][r]));
        #pragma unroll
        for (int off = 1; off < 16; off <<= 1) m = fmaxf(m, __shfl_xor(m, off, 32));
        mnew[r] = fmaxf(rmax[r], m);
      }
      float corr[8], psum[8];
      #pragma unroll
      for (int r = 0; r < 8; ++r) {
        corr[r] = __expf(rmax[r] - mnew[r]);
        rmax[r] = mnew[r];
        psum[r] = 0.f;
      }
      #pragma unroll
      for (int t = 0; t < 8; ++t)
        #pragma unroll
        for (int r = 0; r < 8; ++r) o[t][r] *= corr[r];
      // P = exp(S - m), write per-wave P tile to LDS in [q][key] for A-fragments
      #pragma unroll
      for (int nt = 0; nt < 4; ++nt)
        #pragma unroll
        for (int r = 0; r < 8; ++r) {
          float pv = __expf(s4[nt][r] - rmax[r]);
          psum[r] += pv;
          sPw[(r + 8 * kh) * LDP + nt * 16 + m16] = (bf16_t)pv;
        }
      #pragma unroll
      for (int r = 0; r < 8; ++r) {
        float ps = psum[r];
        #pragma unroll
        for (int off = 1; off < 16; off <<= 1) ps += __shfl_xor(ps, off, 32);
        rsum[r] = rsum[r] * corr[r] + ps;
      }
      __builtin_amdgcn_wave_barrier();   // pin ds_store(P) before ds_load(P) ordering
      // O += P V  (2 k-chunks x 8 d-tiles)
      #pragma unroll
      for (int c = 0; c < 2; ++c) {
        v16bf pf = ld_fragA(&sPw[m16 * LDP + c * 32], kh);
        #pragma unroll
        for (int t = 0; t < 8; ++t) {
          v16bf bv = ld_fragB(&sV[(t * 16 + m16) * LDV + c * 32], kh);
          o[t] = WMMA_BF16(pf, bv, o[t]);
        }
      }
    }
    __syncthreads();
  }
  // epilogue: normalize and store bf16
  #pragma unroll
  for (int t = 0; t < 8; ++t)
    #pragma unroll
    for (int r = 0; r < 8; ++r) {
      int qg = qbase + r + 8 * kh;
      float val = o[t][r] / rsum[r];
      Ob[(size_t)qg * (NH * HD) + h * HD + t * 16 + m16] = (bf16_t)val;
    }
}

extern "C" void kernel_launch(void* const* d_in, const int* in_sizes, int n_in,
                              void* d_out, int out_size, void* d_ws, size_t ws_size,
                              hipStream_t stream) {
  (void)in_sizes; (void)n_in; (void)out_size; (void)ws_size;
  const float* hs   = (const float*)d_in[0];
  // d_in[1] = attention_mask: unused — causal mask computed arithmetically (saves 16MB read)
  const float* wq   = (const float*)d_in[2];
  const float* wk   = (const float*)d_in[3];
  const float* wv   = (const float*)d_in[4];
  const float* wo   = (const float*)d_in[5];
  const float* cosT = (const float*)d_in[6];
  const float* sinT = (const float*)d_in[7];
  float* out = (float*)d_out;

  char* p = (char*)d_ws;
  auto carve = [&](size_t bytes) -> void* {
    void* r = (void*)p; p += (bytes + 255) & ~(size_t)255; return r;
  };
  const size_t SQ = (size_t)S_LEN * HID_DIM;      // 2048*4096
  const size_t SK = (size_t)S_LEN * NKV * HD;     // 2048*1024
  bf16_t* Xb  = (bf16_t*)carve(SQ * 2);                               // [S][HID]
  bf16_t* WQt = (bf16_t*)carve((size_t)HID_DIM * NH * HD * 2);        // [N][K]
  bf16_t* WKt = (bf16_t*)carve((size_t)HID_DIM * NKV * HD * 2);
  bf16_t* WVt = (bf16_t*)carve((size_t)HID_DIM * NKV * HD * 2);
  bf16_t* WOt = (bf16_t*)carve((size_t)NH * HD * HID_DIM * 2);
  float*  Qf  = (float*)carve(SQ * 4);
  float*  Kf  = (float*)carve(SK * 4);
  float*  Vf  = (float*)carve(SK * 4);
  bf16_t* Qbf = (bf16_t*)carve(SQ * 2);
  bf16_t* Kbf = (bf16_t*)carve(SK * 2);
  bf16_t* Vtb = (bf16_t*)carve(SK * 2);                               // [NKV*HD][S]
  bf16_t* Obf = (bf16_t*)carve(SQ * 2);

  const int TB = 256;
  // 1) activations: plain convert; weights: transpose-convert to [N][K]
  cvt_f32_bf16<<<(int)(SQ / 4 / TB), TB, 0, stream>>>(hs, Xb, (int)(SQ / 4));
  cvt_t_f32_bf16<<<dim3(NH * HD / 32, HID_DIM / 32), TB, 0, stream>>>(wq, WQt, HID_DIM, NH * HD);
  cvt_t_f32_bf16<<<dim3(NKV * HD / 32, HID_DIM / 32), TB, 0, stream>>>(wk, WKt, HID_DIM, NKV * HD);
  cvt_t_f32_bf16<<<dim3(NKV * HD / 32, HID_DIM / 32), TB, 0, stream>>>(wv, WVt, HID_DIM, NKV * HD);
  cvt_t_f32_bf16<<<dim3(HID_DIM / 32, NH * HD / 32), TB, 0, stream>>>(wo, WOt, NH * HD, HID_DIM);
  // 2) QKV projections (WMMA bf16, fp32 accum); N divisible by 256 for all
  gemm_bf16_f32<<<dim3(NH * HD / 256, S_LEN / 128), TB, 0, stream>>>(Xb, WQt, Qf, S_LEN, NH * HD, HID_DIM);
  gemm_bf16_f32<<<dim3(NKV * HD / 256, S_LEN / 128), TB, 0, stream>>>(Xb, WKt, Kf, S_LEN, NKV * HD, HID_DIM);
  gemm_bf16_f32<<<dim3(NKV * HD / 256, S_LEN / 128), TB, 0, stream>>>(Xb, WVt, Vf, S_LEN, NKV * HD, HID_DIM);
  // 3) RoPE (Q,K) + bf16; V: transpose-convert to [NKV*HD][S]
  rope_to_bf16<<<S_LEN * NH * 64 / TB, TB, 0, stream>>>(Qf, cosT, sinT, Qbf, NH);
  rope_to_bf16<<<S_LEN * NKV * 64 / TB, TB, 0, stream>>>(Kf, cosT, sinT, Kbf, NKV);
  cvt_t_f32_bf16<<<dim3(NKV * HD / 32, S_LEN / 32), TB, 0, stream>>>(Vf, Vtb, S_LEN, NKV * HD);
  // 4) causal flash attention
  attn_fwd<<<dim3(S_LEN / 128, NH), TB, 0, stream>>>(Qbf, Kbf, Vtb, Obf);
  // 5) output projection -> fp32 d_out
  gemm_bf16_f32<<<dim3(HID_DIM / 256, S_LEN / 128), TB, 0, stream>>>(Obf, WOt, out, S_LEN, HID_DIM, NH * HD);
}
